// QRNN_73873437491668
// MI455X (gfx1250) — matrix operand order repaired and checked by reading, not compiled, not run on hardware
//
#include <hip/hip_runtime.h>
#include <hip/hip_bf16.h>
#include <math.h>

typedef __attribute__((ext_vector_type(16))) __bf16 v16bf;
typedef __attribute__((ext_vector_type(8)))  float  v8f;

#define T_LEN 65536
#define E_DIM 512
#define H_DIM 512
#define N_DIM 1536   // 3H
#define K_DIM 1024   // 2E
#define CH    256    // scan chunk length
#define NCH   (T_LEN / CH)  // 256 chunks

// ---------------------------------------------------------------------------
// Transpose + convert W [2E, 3H] f32 (row-major) -> Wt [3H, 2E] bf16
// so B-matrix columns are contiguous for WMMA fragment loads.
// ---------------------------------------------------------------------------
__global__ __launch_bounds__(256) void wt_convert(const float* __restrict__ W,
                                                  __bf16* __restrict__ Wt) {
    int id = blockIdx.x * 256 + threadIdx.x;        // over K_DIM*N_DIM
    if (id >= K_DIM * N_DIM) return;
    int kk = id & (K_DIM - 1);
    int n  = id >> 10;
    Wt[(size_t)n * K_DIM + kk] = (__bf16)W[(size_t)kk * N_DIM + n];
}

__device__ __forceinline__ float fast_sigmoid(float v) {
    return 1.0f / (1.0f + __expf(-v));
}
__device__ __forceinline__ float fast_tanh(float v) {
    return 2.0f / (1.0f + __expf(-2.0f * v)) - 1.0f;
}

// A fragment: 16x32 bf16 per ISA 7.12.2. Lane holds row m,
// K = {kh*16+0..15} (elems 0..15), contiguous to match B.
// p points at x_row + kh*16 + k.
__device__ __forceinline__ v16bf load_afrag(const float* __restrict__ p) {
    const float4* q = (const float4*)p;
    float4 u0 = q[0], u1 = q[1];
    const float4* r = (const float4*)(p + 8);
    float4 w0 = r[0], w1 = r[1];
    v16bf a;
    a[0]=(__bf16)u0.x;  a[1]=(__bf16)u0.y;  a[2]=(__bf16)u0.z;  a[3]=(__bf16)u0.w;
    a[4]=(__bf16)u1.x;  a[5]=(__bf16)u1.y;  a[6]=(__bf16)u1.z;  a[7]=(__bf16)u1.w;
    a[8]=(__bf16)w0.x;  a[9]=(__bf16)w0.y;  a[10]=(__bf16)w0.z; a[11]=(__bf16)w0.w;
    a[12]=(__bf16)w1.x; a[13]=(__bf16)w1.y; a[14]=(__bf16)w1.z; a[15]=(__bf16)w1.w;
    return a;
}

// ---------------------------------------------------------------------------
// One 16(M) x 128(N) WMMA strip. EDGE=true only for the m0==0 wave
// (t==0 row uses zero-padded x_prev); all other waves get a branch-free loop.
// ---------------------------------------------------------------------------
template<bool EDGE>
__device__ __forceinline__ void gemm_tile(const float* __restrict__ x,
                                          const __bf16* __restrict__ Wt,
                                          const float* __restrict__ bias,
                                          float* __restrict__ G,
                                          int m0, int n0, int lane) {
    const int lr = lane & 15;
    const int kh = lane >> 4;                       // K-half select (contiguous k order, same as B)
    const int m  = m0 + lr;                         // A-row (time index)

    v8f acc[8] = {};

    // B fragment base pointers: column n, K = kh*16 + 0..15 (contiguous 32B)
    const __bf16* bp[8];
    #pragma unroll
    for (int j = 0; j < 8; ++j)
        bp[j] = Wt + (size_t)(n0 + j * 16 + lr) * K_DIM + kh * 16;

    // ---- Phase 1: concat-K [0,512): A rows come from x[t] ----
    {
        const float* ar = x + (size_t)m * E_DIM + kh * 16;
        for (int k = 0; k < E_DIM; k += 32) {
            v16bf a = load_afrag(ar + k);
            #pragma unroll
            for (int j = 0; j < 8; ++j) {
                v16bf bf = *(const v16bf*)(bp[j] + k);
                acc[j] = __builtin_amdgcn_wmma_f32_16x16x32_bf16(
                    false, a, false, bf, (short)0, acc[j], false, false);
            }
        }
    }
    // ---- Phase 2: concat-K [512,1024): A rows come from x[t-1] ----
    {
        const float* ar = x + (size_t)(m - 1) * E_DIM + kh * 16;
        for (int k = 0; k < E_DIM; k += 32) {
            v16bf a;
            if (EDGE && m == 0) {
                #pragma unroll
                for (int i = 0; i < 16; ++i) a[i] = (__bf16)0.0f;
            } else {
                a = load_afrag(ar + k);
            }
            #pragma unroll
            for (int j = 0; j < 8; ++j) {
                v16bf bf = *(const v16bf*)(bp[j] + E_DIM + k);
                acc[j] = __builtin_amdgcn_wmma_f32_16x16x32_bf16(
                    false, a, false, bf, (short)0, acc[j], false, false);
            }
        }
    }

    // Epilogue: bias + activation, write activated gates.
    #pragma unroll
    for (int j = 0; j < 8; ++j) {
        const int n    = n0 + j * 16 + lr;
        const int gate = n >> 9;                    // 0=z(tanh), 1=f(sig), 2=o(sig)
        const float bi = bias[n];
        #pragma unroll
        for (int r = 0; r < 8; ++r) {
            const int t = m0 + kh * 8 + r;          // D layout: lane-half selects row group
            float v = acc[j][r] + bi;
            float act = (gate == 0) ? fast_tanh(v) : fast_sigmoid(v);
            G[(size_t)t * N_DIM + n] = act;
        }
    }
}

// ---------------------------------------------------------------------------
// gates = concat(x_t, x_{t-1}) @ W + b, activations fused, stored to G[T,1536].
// Block = 8 waves; wave computes a 16(M) x 128(N) strip via 8 WMMA accumulators.
// A converted f32->bf16 on the fly; B read as bf16 from Wt (L2-resident, 3MB).
// ---------------------------------------------------------------------------
__global__ __launch_bounds__(256) void gates_gemm(const float* __restrict__ x,
                                                  const __bf16* __restrict__ Wt,
                                                  const float* __restrict__ bias,
                                                  float* __restrict__ G) {
    const int lane = threadIdx.x & 31;
    const int wave = threadIdx.x >> 5;
    const int m0 = blockIdx.x * 128 + wave * 16;    // time-tile base
    const int n0 = blockIdx.y * 128;                // gate-column tile base
    if (m0 == 0)                                    // wave-uniform branch
        gemm_tile<true>(x, Wt, bias, G, m0, n0, lane);
    else
        gemm_tile<false>(x, Wt, bias, G, m0, n0, lane);
}

// ---------------------------------------------------------------------------
// Phase 1 of chunked linear-recurrence scan: per (channel, chunk) compute
// A = prod(f), B = fold of f*B + (1-f)*z over the chunk.
// ---------------------------------------------------------------------------
__global__ __launch_bounds__(256) void chunk_reduce(const float* __restrict__ G,
                                                    float* __restrict__ Asum,
                                                    float* __restrict__ Bsum) {
    int id = blockIdx.x * 256 + threadIdx.x;        // NCH * H_DIM threads
    int h  = id & (H_DIM - 1);
    int ch = id >> 9;
    float A = 1.0f, B = 0.0f;
    int t0 = ch * CH;
    #pragma unroll 4
    for (int s = 0; s < CH; ++s) {
        size_t base = (size_t)(t0 + s) * N_DIM;
        __builtin_prefetch(G + base + 8 * (size_t)N_DIM + h, 0, 0);
        float z = G[base + h];
        float f = G[base + H_DIM + h];
        B = f * B + (1.0f - f) * z;
        A *= f;
    }
    Asum[ch * H_DIM + h] = A;
    Bsum[ch * H_DIM + h] = B;
}

// ---------------------------------------------------------------------------
// Phase 2: exclusive scan over chunk summaries (tiny: 512 channels x 256 chunks)
// ---------------------------------------------------------------------------
__global__ __launch_bounds__(256) void chunk_scan(const float* __restrict__ Asum,
                                                  const float* __restrict__ Bsum,
                                                  float* __restrict__ Cstart) {
    int h = blockIdx.x * 256 + threadIdx.x;
    if (h >= H_DIM) return;
    float c = 0.0f;
    for (int ch = 0; ch < NCH; ++ch) {
        Cstart[ch * H_DIM + h] = c;
        c = Asum[ch * H_DIM + h] * c + Bsum[ch * H_DIM + h];
    }
}

// ---------------------------------------------------------------------------
// Phase 3: apply carry-in per chunk, emit h_t = o_t * c_t, plus final (h, c).
// ---------------------------------------------------------------------------
__global__ __launch_bounds__(256) void fo_apply(const float* __restrict__ G,
                                                const float* __restrict__ Cstart,
                                                float* __restrict__ out) {
    int id = blockIdx.x * 256 + threadIdx.x;        // NCH * H_DIM threads
    int h  = id & (H_DIM - 1);
    int ch = id >> 9;
    float c  = Cstart[ch * H_DIM + h];
    float hv = 0.0f;
    int t0 = ch * CH;
    #pragma unroll 4
    for (int s = 0; s < CH; ++s) {
        int t = t0 + s;
        size_t base = (size_t)t * N_DIM;
        __builtin_prefetch(G + base + 8 * (size_t)N_DIM + h, 0, 0);
        float z = G[base + h];
        float f = G[base + H_DIM + h];
        float o = G[base + 2 * H_DIM + h];
        c  = f * c + (1.0f - f) * z;
        hv = o * c;
        out[(size_t)t * H_DIM + h] = hv;
    }
    if (ch == NCH - 1) {                            // final h and c rows
        out[(size_t)T_LEN * H_DIM + h]         = hv;
        out[(size_t)T_LEN * H_DIM + H_DIM + h] = c;
    }
}

// ---------------------------------------------------------------------------
extern "C" void kernel_launch(void* const* d_in, const int* in_sizes, int n_in,
                              void* d_out, int out_size, void* d_ws, size_t ws_size,
                              hipStream_t stream) {
    const float* x    = (const float*)d_in[0];   // [T, E]
    const float* W    = (const float*)d_in[1];   // [2E, 3H]
    const float* bias = (const float*)d_in[2];   // [3H]
    float* out = (float*)d_out;                  // [T*H + H + H]

    char* w = (char*)d_ws;
    const size_t G_bytes  = (size_t)T_LEN * N_DIM * sizeof(float);     // 384 MB
    const size_t Wt_bytes = (size_t)K_DIM * N_DIM * sizeof(__bf16);    // 3 MB
    float*  G      = (float*)w;
    __bf16* Wt     = (__bf16*)(w + G_bytes);
    float*  Asum   = (float*)(w + G_bytes + Wt_bytes);
    float*  Bsum   = Asum + NCH * H_DIM;
    float*  Cstart = Bsum + NCH * H_DIM;

    wt_convert<<<(K_DIM * N_DIM) / 256, 256, 0, stream>>>(W, Wt);

    dim3 gg(T_LEN / 128, N_DIM / 128);           // 512 x 12 blocks, 8 waves each
    gates_gemm<<<gg, 256, 0, stream>>>(x, Wt, bias, G);

    chunk_reduce<<<(NCH * H_DIM) / 256, 256, 0, stream>>>(G, Asum, Bsum);
    chunk_scan<<<(H_DIM + 255) / 256, 256, 0, stream>>>(Asum, Bsum, Cstart);
    fo_apply<<<(NCH * H_DIM) / 256, 256, 0, stream>>>(G, Cstart, out);
}
